// NN_computeF_3410204033339
// MI455X (gfx1250) — matrix operand order repaired and checked by reading, not compiled, not run on hardware
//
#include <hip/hip_runtime.h>

typedef __attribute__((ext_vector_type(16))) _Float16 v16h;
typedef __attribute__((ext_vector_type(8)))  float    v8f;

static __device__ __forceinline__ v8f wmma16(v16h a, v16h b, v8f c) {
  // D = A(16x32 f16) * B(32x16 f16) + C(16x16 f32)
  return __builtin_amdgcn_wmma_f32_16x16x32_f16(false, a, false, b, (short)0, c, false, false);
}

static __device__ __forceinline__ v16h load_b16(const _Float16* p) {
  return *(const v16h*)p;  // 32B contiguous per lane -> global_load_b128 x2
}

// Gather the A-operand (16x32 f16) for this lane from an f16 activation array.
// baseHalf already includes row*stride + kc*32 + (lane>=16 ? 8 : 0).
static __device__ __forceinline__ v16h load_a16(const _Float16* base, int baseHalf) {
  union { v16h v; unsigned u[8]; } a;
  const unsigned* up = (const unsigned*)base;
#pragma unroll
  for (int j = 0; j < 8; ++j) {
    const int kb = (j < 4) ? (2 * j) : (2 * j + 8);
    a.u[j] = up[(baseHalf + kb) >> 1];
  }
  return a.v;
}

static __device__ __forceinline__ float relu(float x) { return x > 0.f ? x : 0.f; }

// Pack row-major Linear weight W[N][K] (torch layout) into WMMA-B tile order:
// packed[((ct*nK + kc)*32 + lane)*16 + h] = W[ct*16 + lane%16][kc*32 + (lane/16)*8 + h + (h/8)*8]
static __device__ void pack_b(const float* __restrict__ w, _Float16* __restrict__ dst,
                              int N, int K, int tid, int nth) {
  const int total = N * K;
  const int nK = K >> 5;
  for (int idx = tid; idx < total; idx += nth) {
    const int h    = idx & 15;
    const int lane = (idx >> 4) & 31;
    const int cc   = idx >> 9;          // ct*nK + kc
    const int kc   = cc % nK;
    const int ct   = cc / nK;
    const int nn   = ct * 16 + (lane & 15);
    const int kk   = kc * 32 + ((lane >> 4) << 3) + h + ((h >> 3) << 3);
    dst[idx] = (_Float16)w[nn * K + kk];
  }
}

__global__ void prep_kernel(const float* __restrict__ V, float* __restrict__ logV,
                            const float* w2a, const float* w3a, const float* w4a,
                            const float* w2b, const float* w3b, const float* w4b,
                            _Float16* p1l2, _Float16* p1l3, _Float16* p1l4,
                            _Float16* p2l2, _Float16* p2l3, _Float16* p2l4, int n) {
  const int tid = blockIdx.x * blockDim.x + threadIdx.x;
  const int nth = blockDim.x * gridDim.x;
  for (int i = tid; i < n; i += nth) logV[i] = logf(V[i]);
  pack_b(w2a, p1l2, 512, 128, tid, nth);
  pack_b(w3a, p1l3, 512, 512, tid, nth);
  pack_b(w4a, p1l4, 128, 512, tid, nth);
  pack_b(w2b, p2l2, 512, 128, tid, nth);
  pack_b(w3b, p2l3, 512, 512, tid, nth);
  pack_b(w4b, p2l4, 128, 512, tid, nth);
}

// ---------------- Sequential Euler scan: 1 workgroup, 16 waves ----------------
__global__ void __launch_bounds__(512)
scan_kernel(const float* __restrict__ V, const float* __restrict__ logV,
            const float* __restrict__ w1, const float* __restrict__ b1,
            const _Float16* __restrict__ pl2, const float* __restrict__ b2,
            const _Float16* __restrict__ pl3, const float* __restrict__ b3,
            const _Float16* __restrict__ pl4, const float* __restrict__ b4,
            const float* __restrict__ w5, const float* __restrict__ b5,
            float* __restrict__ Xi, int n, float dt) {
  __shared__ __align__(16) _Float16 h1[128];
  __shared__ __align__(16) _Float16 h2[512];
  __shared__ __align__(16) _Float16 h3[512];
  __shared__ float h4p[2][128];
  __shared__ float s_xi, s_exi;
  __shared__ float s_w1[512], s_b1[128], s_b2[512], s_b3[512], s_b4[128], s_w5[128];
  __shared__ float s_b5;

  const int tid  = threadIdx.x;
  const int lane = tid & 31;
  const int wave = tid >> 5;
  const int agrp = (lane >> 4) << 3;     // +8 halves for lanes 16..31

  // Preload all small NN2 parameters to LDS once.
  if (tid < 512) { s_w1[tid] = w1[tid]; s_b2[tid] = b2[tid]; s_b3[tid] = b3[tid]; }
  if (tid < 128) { s_b1[tid] = b1[tid]; s_b4[tid] = b4[tid]; s_w5[tid] = w5[tid]; }
  if (tid == 0) { s_b5 = b5[0]; s_xi = 0.f; s_exi = 1.f; Xi[0] = 0.f; }
  __syncthreads();

  int z = 0;   // opaque zero: keeps weight-load addresses loop-variant (no LICM
               // hoist/spill) while preserving global address-space inference.

  for (int i = 1; i < n; ++i) {
    asm volatile("" : "+s"(z));
    const _Float16* l2p = pl2 + z;
    const _Float16* l3p = pl3 + z;
    const _Float16* l4p = pl4 + z;

    const float xi = s_xi, exi = s_exi;
    if (tid < 128) {                       // layer1: 4 -> 128
      const float v = V[i], lv = logV[i];
      h1[tid] = (_Float16)relu(s_b1[tid] + s_w1[tid*4]*v + s_w1[tid*4+1]*lv
                               + s_w1[tid*4+2]*xi + s_w1[tid*4+3]*exi);
    }
    __syncthreads();

    {                                      // layer2: 128 -> 512, 2 tiles/wave
      v8f acc0 = {}, acc1 = {};
      const int base = ((wave * 2) * 4) * 512;   // ((ct0*4+0)*32+0)<<4
#pragma unroll
      for (int kc = 0; kc < 4; ++kc) {
        v16h a  = load_a16(h1, kc * 32 + agrp);
        v16h b0 = load_b16(l2p + base + ((kc * 32 + lane) << 4));
        v16h b1v = load_b16(l2p + base + 2048 + ((kc * 32 + lane) << 4));
        acc0 = wmma16(a, b0, acc0);
        acc1 = wmma16(a, b1v, acc1);
      }
      if (lane < 16) {
        const int n0 = wave * 32 + lane;
        h2[n0]      = (_Float16)relu(acc0[0] + s_b2[n0]);
        h2[n0 + 16] = (_Float16)relu(acc1[0] + s_b2[n0 + 16]);
      }
    }
    __syncthreads();

    {                                      // layer3: 512 -> 512, 2 tiles/wave
      v8f acc0 = {}, acc1 = {};
      const int base = ((wave * 2) * 16) * 512;  // ((ct0*16+0)*32+0)<<4
#pragma unroll
      for (int kc = 0; kc < 16; ++kc) {
        v16h a  = load_a16(h2, kc * 32 + agrp);
        v16h b0 = load_b16(l3p + base + ((kc * 32 + lane) << 4));
        v16h b1v = load_b16(l3p + base + 8192 + ((kc * 32 + lane) << 4));
        acc0 = wmma16(a, b0, acc0);
        acc1 = wmma16(a, b1v, acc1);
      }
      if (lane < 16) {
        const int n0 = wave * 32 + lane;
        h3[n0]      = (_Float16)relu(acc0[0] + s_b3[n0]);
        h3[n0 + 16] = (_Float16)relu(acc1[0] + s_b3[n0 + 16]);
      }
    }
    __syncthreads();

    {                                      // layer4: 512 -> 128, K split over wave pairs
      const int ct = wave >> 1;
      const int kh = wave & 1;
      v8f acc = {};
#pragma unroll
      for (int kc = 0; kc < 8; ++kc) {
        const int kcc = kh * 8 + kc;
        v16h a = load_a16(h3, kcc * 32 + agrp);
        v16h b = load_b16(l4p + (((ct * 16 + kcc) * 32 + lane) << 4));
        acc = wmma16(a, b, acc);
      }
      if (lane < 16) h4p[kh][ct * 16 + lane] = acc[0];
    }
    __syncthreads();

    if (wave == 0) {                       // layer5: 128 -> 1, Euler update
      float p = 0.f;
#pragma unroll
      for (int j = 0; j < 4; ++j) {
        const int idx = lane + 32 * j;
        const float hv = relu(s_b4[idx] + h4p[0][idx] + h4p[1][idx]);
        p += s_w5[idx] * hv;
      }
#pragma unroll
      for (int d = 16; d > 0; d >>= 1) p += __shfl_down(p, d, 32);
      if (lane == 0) {
        const float xin = xi + (p + s_b5) * dt;
        s_xi = xin;
        s_exi = expf(xin);
        Xi[i] = xin;
      }
    }
    __syncthreads();
  }
}

// ---------------- Batched NN1 pass: real 16x16 WMMA tiles ----------------
#define BROWS 16
#define BSTRIDE 520   // 512 + 8-half pad -> 4-bank rotation per row

__global__ void __launch_bounds__(256)
nn1_kernel(const float* __restrict__ V, const float* __restrict__ logV,
           const float* __restrict__ Xi,
           const float* __restrict__ w1, const float* __restrict__ b1,
           const _Float16* __restrict__ pl2, const float* __restrict__ b2,
           const _Float16* __restrict__ pl3, const float* __restrict__ b3,
           const _Float16* __restrict__ pl4, const float* __restrict__ b4,
           const float* __restrict__ w5, const float* __restrict__ b5,
           float* __restrict__ out, int n) {
  __shared__ __align__(16) _Float16 bufA[BROWS * BSTRIDE];
  __shared__ __align__(16) _Float16 bufB[BROWS * BSTRIDE];
  __shared__ float sv[BROWS], slv[BROWS], sxi[BROWS], sexi[BROWS];
  __shared__ float red[256];

  const int tid  = threadIdx.x;
  const int lane = tid & 31;
  const int wave = tid >> 5;
  const int agrp = (lane >> 4) << 3;
  const int rowBase = (lane & 15) * BSTRIDE;
  const int row0 = blockIdx.x * BROWS;

  if (tid < BROWS) {
    int g = row0 + tid;
    if (g >= n) g = n - 1;
    const float xv = Xi[g];
    sv[tid] = V[g]; slv[tid] = logV[g]; sxi[tid] = xv; sexi[tid] = expf(xv);
  }
  __syncthreads();

#pragma unroll
  for (int e = 0; e < (BROWS * 128) / 256; ++e) {   // layer1: 16x128
    const int idx = tid + 256 * e;
    const int r = idx >> 7, c = idx & 127;
    const float* w = w1 + c * 4;
    bufA[r * BSTRIDE + c] =
        (_Float16)relu(b1[c] + w[0]*sv[r] + w[1]*slv[r] + w[2]*sxi[r] + w[3]*sexi[r]);
  }
  __syncthreads();

  {                                                 // layer2: K=128 N=512, 4 tiles/wave
    v8f acc[4] = {v8f{}, v8f{}, v8f{}, v8f{}};
#pragma unroll
    for (int kc = 0; kc < 4; ++kc) {
      v16h a = load_a16(bufA, rowBase + kc * 32 + agrp);
#pragma unroll
      for (int t = 0; t < 4; ++t) {
        const int ct = wave + 8 * t;
        v16h b = load_b16(pl2 + (((ct * 4 + kc) * 32 + lane) << 4));
        acc[t] = wmma16(a, b, acc[t]);
      }
    }
#pragma unroll
    for (int t = 0; t < 4; ++t) {
      const int nn = (wave + 8 * t) * 16 + (lane & 15);
      const float bv = b2[nn];
#pragma unroll
      for (int r = 0; r < 8; ++r)
        bufB[(agrp + r) * BSTRIDE + nn] = (_Float16)relu(acc[t][r] + bv);
    }
  }
  __syncthreads();

  {                                                 // layer3: K=512 N=512, 4 tiles/wave
    v8f acc[4] = {v8f{}, v8f{}, v8f{}, v8f{}};
#pragma unroll
    for (int kc = 0; kc < 16; ++kc) {
      v16h a = load_a16(bufB, rowBase + kc * 32 + agrp);
#pragma unroll
      for (int t = 0; t < 4; ++t) {
        const int ct = wave + 8 * t;
        v16h b = load_b16(pl3 + (((ct * 16 + kc) * 32 + lane) << 4));
        acc[t] = wmma16(a, b, acc[t]);
      }
    }
#pragma unroll
    for (int t = 0; t < 4; ++t) {
      const int nn = (wave + 8 * t) * 16 + (lane & 15);
      const float bv = b3[nn];
#pragma unroll
      for (int r = 0; r < 8; ++r)
        bufA[(agrp + r) * BSTRIDE + nn] = (_Float16)relu(acc[t][r] + bv);
    }
  }
  __syncthreads();

  {                                                 // layer4: K=512 N=128, 1 tile/wave
    const int ct = wave;
    v8f acc = {};
#pragma unroll
    for (int kc = 0; kc < 16; ++kc) {
      v16h a = load_a16(bufA, rowBase + kc * 32 + agrp);
      v16h b = load_b16(pl4 + (((ct * 16 + kc) * 32 + lane) << 4));
      acc = wmma16(a, b, acc);
    }
    const int nn = ct * 16 + (lane & 15);
    const float bv = b4[nn];
#pragma unroll
    for (int r = 0; r < 8; ++r)
      bufB[(agrp + r) * BSTRIDE + nn] = (_Float16)relu(acc[r] + bv);
  }
  __syncthreads();

  {                                                 // layer5: 128 -> 1 (+F0)
    const int r = tid >> 4;
    const int q = tid & 15;
    float p = 0.f;
#pragma unroll
    for (int m = 0; m < 8; ++m)
      p += w5[q * 8 + m] * (float)bufB[r * BSTRIDE + q * 8 + m];
    red[tid] = p;
    __syncthreads();
    if (tid < BROWS) {
      float s = 0.f;
#pragma unroll
      for (int j = 0; j < 16; ++j) s += red[tid * 16 + j];
      const int g = row0 + tid;
      if (g < n) out[g] = 0.58f + b5[0] + s;
    }
  }
}

extern "C" void kernel_launch(void* const* d_in, const int* in_sizes, int n_in,
                              void* d_out, int out_size, void* d_ws, size_t ws_size,
                              hipStream_t stream) {
  (void)n_in; (void)out_size; (void)ws_size;
  const float* V    = (const float*)d_in[0];
  const float* n1w1 = (const float*)d_in[1];
  const float* n1b1 = (const float*)d_in[2];
  const float* n1w2 = (const float*)d_in[3];
  const float* n1b2 = (const float*)d_in[4];
  const float* n1w3 = (const float*)d_in[5];
  const float* n1b3 = (const float*)d_in[6];
  const float* n1w4 = (const float*)d_in[7];
  const float* n1b4 = (const float*)d_in[8];
  const float* n1w5 = (const float*)d_in[9];
  const float* n1b5 = (const float*)d_in[10];
  const float* n2w1 = (const float*)d_in[11];
  const float* n2b1 = (const float*)d_in[12];
  const float* n2w2 = (const float*)d_in[13];
  const float* n2b2 = (const float*)d_in[14];
  const float* n2w3 = (const float*)d_in[15];
  const float* n2b3 = (const float*)d_in[16];
  const float* n2w4 = (const float*)d_in[17];
  const float* n2b4 = (const float*)d_in[18];
  const float* n2w5 = (const float*)d_in[19];
  const float* n2b5 = (const float*)d_in[20];

  const int n = in_sizes[0];
  char* ws = (char*)d_ws;
  size_t off = 0;
  auto take = [&](size_t bytes) { size_t r = off; off = (off + bytes + 255) & ~(size_t)255; return r; };
  float*    logV = (float*)   (ws + take((size_t)n * 4));
  float*    Xi   = (float*)   (ws + take((size_t)n * 4));
  _Float16* p1l2 = (_Float16*)(ws + take(512 * 128 * 2));
  _Float16* p1l3 = (_Float16*)(ws + take(512 * 512 * 2));
  _Float16* p1l4 = (_Float16*)(ws + take(128 * 512 * 2));
  _Float16* p2l2 = (_Float16*)(ws + take(512 * 128 * 2));
  _Float16* p2l3 = (_Float16*)(ws + take(512 * 512 * 2));
  _Float16* p2l4 = (_Float16*)(ws + take(128 * 512 * 2));

  const float dt = 30.0f / (float)(n - 1);

  hipLaunchKernelGGL(prep_kernel, dim3(256), dim3(256), 0, stream,
                     V, logV, n1w2, n1w3, n1w4, n2w2, n2w3, n2w4,
                     p1l2, p1l3, p1l4, p2l2, p2l3, p2l4, n);
  hipLaunchKernelGGL(scan_kernel, dim3(1), dim3(512), 0, stream,
                     V, logV, n2w1, n2b1, p2l2, n2b2, p2l3, n2b3, p2l4, n2b4,
                     n2w5, n2b5, Xi, n, dt);
  hipLaunchKernelGGL(nn1_kernel, dim3((n + BROWS - 1) / BROWS), dim3(256), 0, stream,
                     V, logV, Xi, n1w1, n1b1, p1l2, n1b2, p1l3, n1b3, p1l4, n1b4,
                     n1w5, n1b5, (float*)d_out, n);
}
